// FastLinearMemory_15453292331558
// MI455X (gfx1250) — compile-verified, matrix operand order checked by reading, not run on hardware
//
#include <hip/hip_runtime.h>

// ---------------- problem sizes (fixed by reference) ----------------
#define N_TOK  8192
#define M_MEM  128
#define D_DIM  64
#define H_HEADS 8
#define DMODEL 1024
#define HD     512
#define EPSV   1e-5f

// ---------------- bf16 helpers (hardware convert via __bf16) ----------------
typedef __attribute__((ext_vector_type(16))) __bf16 v16bf;
typedef __attribute__((ext_vector_type(8)))  float  v8f;

__device__ __forceinline__ unsigned short f2bf(float f) {
  union { __bf16 b; unsigned short u; } cv;
  cv.b = (__bf16)f;                       // hardware f32->bf16 (RNE)
  return cv.u;
}
__device__ __forceinline__ unsigned packbf(float lo, float hi) {
  union { __bf16 b[2]; unsigned u; } cv;
  cv.b[0] = (__bf16)lo;
  cv.b[1] = (__bf16)hi;
  return cv.u;                            // lowers to v_cvt_pk_bf16_f32
}
__device__ __forceinline__ float bf2f(unsigned v) {   // low 16 bits hold bf16
  return __uint_as_float(v << 16);
}

union FragU { v16bf v; unsigned u[8]; };

// Load a 16x32 bf16 fragment (A layout; Bt rows use the mirrored layout with
// row==column-of-B) from a row-major LDS tile with leading dim `ld` elements.
// ISA 7.12.2: lanes 0-15 rows M=0..15 hold K=0..7 (v0..3) and K=16..23 (v4..7);
// lanes 16-31 hold K=8..15 / K=24..31. Each dword packs a contiguous k-pair.
__device__ __forceinline__ v16bf load_frag(const unsigned short* tile, int ld) {
  const int lane  = threadIdx.x & 31;
  const int r     = lane & 15;
  const int khalf = (lane >> 4) << 3;       // 0 or 8
  const unsigned short* p = tile + r * ld;
  FragU f;
#pragma unroll
  for (int v = 0; v < 8; ++v) {
    const int k = ((v & 4) << 2) + khalf + ((v & 3) << 1);
    f.u[v] = *(const unsigned*)(p + k);
  }
  return f.v;
}

__device__ __forceinline__ v8f wmma_bf16(v16bf a, v16bf b, v8f c) {
  return __builtin_amdgcn_wmma_f32_16x16x32_bf16(
      /*neg_a=*/false, a, /*neg_b=*/false, b,
      /*c_mod=*/(short)0, c, /*reuse_a=*/false, /*reuse_b=*/false);
}

// ---------------- async global->LDS copy (CDNA5, ASYNCcnt tracked) ----------
__device__ __forceinline__ void async_copy_b128(void* lds_dst, const void* gsrc) {
  // LDS aperture flat address: low 32 bits are the LDS byte offset (ISA 10.2).
  unsigned lds_off = (unsigned)(uintptr_t)lds_dst;
  unsigned long long ga = (unsigned long long)(uintptr_t)gsrc;
  asm volatile("global_load_async_to_lds_b128 %0, %1, off"
               :: "v"(lds_off), "v"(ga) : "memory");
}
template<int N>
__device__ __forceinline__ void wait_async() {
#if __has_builtin(__builtin_amdgcn_s_wait_asynccnt)
  __builtin_amdgcn_s_wait_asynccnt((unsigned short)N);
#else
  asm volatile("s_wait_asynccnt %0" :: "i"(N) : "memory");
#endif
}

// ---------------- weight convert + transpose (W[R,C] fp32 -> Wt[C,R] bf16) ----
__global__ __launch_bounds__(256)
void cvt_transpose(const float* __restrict__ W, unsigned short* __restrict__ Wt,
                   int R, int C) {
  int o = blockIdx.x * 256 + threadIdx.x;
  if (o >= R * C) return;
  int c = o / R;
  int r = o - c * R;
  Wt[o] = f2bf(W[(size_t)r * C + c]);       // coalesced writes, strided reads
}

// ---------------- tiled WMMA GEMM: Out = act(A @ Bt^T + bias) ---------------
// A: M x K (fp32 or bf16), Bt: N x K bf16 row-major (i.e. B^T), block 64x64,
// 128 threads = 4 waves; wave w owns rows [16w,16w+16) x 4 col-tiles.
// Double-buffered: stage s+1 is issued (async) before waiting on stage s;
// ASYNCcnt completes in order, so s_wait_asynccnt <= CPT proves stage s landed.
template<bool AFP32>
__device__ __forceinline__ void stage_tiles(
    const void* Aptr, const unsigned short* Bt,
    unsigned short* As, unsigned short* Bs,
    int row0, int col0, int K, int kk, int lrow, int kh16) {
  if constexpr (AFP32) {
    const float* Ag = (const float*)Aptr + (size_t)(row0 + lrow) * K + kk + kh16;
    unsigned* dst = (unsigned*)(As + lrow * 32 + kh16);
#pragma unroll
    for (int j = 0; j < 16; j += 4) {
      float4 v4 = *(const float4*)(Ag + j);
      dst[(j >> 1) + 0] = packbf(v4.x, v4.y);
      dst[(j >> 1) + 1] = packbf(v4.z, v4.w);
    }
  } else {
    const unsigned short* Ag =
        (const unsigned short*)Aptr + (size_t)(row0 + lrow) * K + kk + kh16;
    unsigned short* dst = As + lrow * 32 + kh16;
    async_copy_b128(dst, Ag);
    async_copy_b128(dst + 8, Ag + 8);
  }
  const unsigned short* Bg = Bt + (size_t)(col0 + lrow) * K + kk + kh16;
  unsigned short* dstB = Bs + lrow * 32 + kh16;
  async_copy_b128(dstB, Bg);
  async_copy_b128(dstB + 8, Bg + 8);
}

template<bool AFP32, bool RELU, bool OUTBF16>
__global__ __launch_bounds__(128)
void gemm_bf16(const void* __restrict__ Aptr, const unsigned short* __restrict__ Bt,
               const float* __restrict__ bias, void* __restrict__ Outp,
               int M, int N, int K) {
  __shared__ unsigned short As[2][64 * 32];
  __shared__ unsigned short Bs[2][64 * 32];
  const int tid  = threadIdx.x;
  const int lane = tid & 31;
  const int wave = tid >> 5;
  const int row0 = blockIdx.x * 64;
  const int col0 = blockIdx.y * 64;
  const int lrow = tid >> 1;              // 0..63 staging row
  const int kh16 = (tid & 1) * 16;        // staging k-half

  constexpr int CPT = AFP32 ? 2 : 4;      // async ops per stage per thread
  const int nsteps = K / 32;

  v8f acc[4] = {};

  stage_tiles<AFP32>(Aptr, Bt, As[0], Bs[0], row0, col0, K, 0, lrow, kh16);
  for (int s = 0; s < nsteps; ++s) {
    const int cur = s & 1;
    if (s + 1 < nsteps) {
      stage_tiles<AFP32>(Aptr, Bt, As[cur ^ 1], Bs[cur ^ 1],
                         row0, col0, K, (s + 1) * 32, lrow, kh16);
      wait_async<CPT>();                  // stage s done, s+1 still in flight
    } else {
      wait_async<0>();
    }
    __syncthreads();
    v16bf a = load_frag(As[cur] + wave * 16 * 32, 32);
#pragma unroll
    for (int ct = 0; ct < 4; ++ct) {
      v16bf b = load_frag(Bs[cur] + ct * 16 * 32, 32);
      acc[ct] = wmma_bf16(a, b, acc[ct]);
    }
    __syncthreads();                      // protect buf before next overwrite
  }

  const int rhi = (lane & 16) ? 8 : 0;
  const int cl  = lane & 15;
#pragma unroll
  for (int ct = 0; ct < 4; ++ct) {
    const int col = col0 + ct * 16 + cl;
    const float bv = bias[col];
#pragma unroll
    for (int i = 0; i < 8; ++i) {
      const int row = row0 + wave * 16 + i + rhi;
      float v = acc[ct][i] + bv;
      if (RELU) v = fmaxf(v, 0.0f);
      if (OUTBF16) ((unsigned short*)Outp)[(size_t)row * N + col] = f2bf(v);
      else         ((float*)Outp)[(size_t)row * N + col] = v;
    }
  }
}

// ---------------- per-token fused kernel ------------------------------------
// attn[n] = ((q kᵀ) mem) / (rowsum(q kᵀ) + eps), reassociated form.
// 1 workgroup (4 waves) per token.  sᵀ = k·qᵀ (128x16,K=64), numᵀ = memᵀ·sᵀ
// (64x16,K=128).  Heads padded 8->16 with zeros so EXEC stays all-ones.
__global__ __launch_bounds__(128)
void token_attn(const float* __restrict__ mem, const float* __restrict__ addr,
                const unsigned short* __restrict__ qws,
                unsigned short* __restrict__ attnws) {
  __shared__ unsigned short kT[M_MEM * D_DIM];   // k[m][d]        16 KB
  __shared__ unsigned short mT[D_DIM * M_MEM];   // mem^T[d][m]    16 KB
  __shared__ unsigned short qT[16 * D_DIM];      // q[head][d]      2 KB
  __shared__ unsigned short sT[16 * M_MEM];      // s[head][m]      4 KB
  __shared__ float denomL[H_HEADS];

  const int n    = blockIdx.x;
  const int tid  = threadIdx.x;
  const int lane = tid & 31;
  const int wave = tid >> 5;

  // Stage memories: each thread owns a row PAIR (m0, m0+1) and half the d
  // range, so mem^T stores pack two adjacent-m bf16 into one aligned b32
  // (stride 4B across lanes -> bank-conflict free).
  {
    const int m0 = (tid & 63) * 2;
    const int d0 = (tid >> 6) * 32;            // 0 or 32
    const float* mA = mem + ((size_t)n * M_MEM + m0) * D_DIM + d0;
    const float* mB = mA + D_DIM;
    const float* aA = addr + (size_t)m0 * D_DIM + d0;
    const float* aB = aA + D_DIM;
    unsigned* kA = (unsigned*)(kT + m0 * D_DIM + d0);
    unsigned* kB = (unsigned*)(kT + (m0 + 1) * D_DIM + d0);
#pragma unroll 2
    for (int d = 0; d < 32; d += 4) {
      float4 va = *(const float4*)(mA + d);
      float4 vb = *(const float4*)(mB + d);
      float4 sa = *(const float4*)(aA + d);
      float4 sb = *(const float4*)(aB + d);
      kA[(d >> 1) + 0] = packbf(fmaxf(va.x + sa.x, 0.f), fmaxf(va.y + sa.y, 0.f));
      kA[(d >> 1) + 1] = packbf(fmaxf(va.z + sa.z, 0.f), fmaxf(va.w + sa.w, 0.f));
      kB[(d >> 1) + 0] = packbf(fmaxf(vb.x + sb.x, 0.f), fmaxf(vb.y + sb.y, 0.f));
      kB[(d >> 1) + 1] = packbf(fmaxf(vb.z + sb.z, 0.f), fmaxf(vb.w + sb.w, 0.f));
      *(unsigned*)(mT + (d0 + d + 0) * M_MEM + m0) = packbf(va.x, vb.x);
      *(unsigned*)(mT + (d0 + d + 1) * M_MEM + m0) = packbf(va.y, vb.y);
      *(unsigned*)(mT + (d0 + d + 2) * M_MEM + m0) = packbf(va.z, vb.z);
      *(unsigned*)(mT + (d0 + d + 3) * M_MEM + m0) = packbf(va.w, vb.w);
    }
  }
  // Stage q (bf16, already relu'd) into rows 0..7; zero pad rows 8..15.
  {
    const int head = tid >> 4;            // 0..7
    const int d0   = (tid & 15) * 4;
    const unsigned* src = (const unsigned*)(qws + (size_t)n * HD + head * D_DIM + d0);
    unsigned* dq = (unsigned*)(qT + head * D_DIM + d0);
    dq[0] = src[0]; dq[1] = src[1];
    unsigned* dz = (unsigned*)(qT + (8 + head) * D_DIM + d0);
    dz[0] = 0u; dz[1] = 0u;
  }
  if (tid < H_HEADS) denomL[tid] = EPSV;
  __syncthreads();

  // GEMM1: sT tiles (two 16-row m-tiles per wave), K=64 -> 2 wmma each.
#pragma unroll
  for (int t = 0; t < 2; ++t) {
    const int rt = wave * 2 + t;
    v8f acc = {};
#pragma unroll
    for (int ks = 0; ks < 2; ++ks) {
      v16bf a = load_frag(kT + rt * 16 * D_DIM + ks * 32, D_DIM);
      v16bf b = load_frag(qT + ks * 32, D_DIM);
      acc = wmma_bf16(a, b, acc);
    }
    const int m0   = rt * 16 + ((lane & 16) ? 8 : 0);
    const int head = lane & 15;           // rows 8..15 get zeros (q pad)
#pragma unroll
    for (int i = 0; i < 8; ++i)
      sT[head * M_MEM + m0 + i] = f2bf(acc[i]);
  }
  __syncthreads();

  // denominator[h] = eps + sum_m s[h][m]   (16 partial sums per head)
  {
    const int head = tid & 7;
    const unsigned* p = (const unsigned*)(sT + head * M_MEM + (tid >> 3) * 8);
    float s = 0.f;
#pragma unroll
    for (int j = 0; j < 4; ++j) {
      const unsigned w = p[j];
      s += bf2f(w & 0xffffu) + bf2f(w >> 16);
    }
    atomicAdd(&denomL[head], s);
  }
  __syncthreads();

  // GEMM2: numT (one 16-row d-tile per wave), K=128 -> 4 wmma each.
  {
    v8f acc = {};
#pragma unroll
    for (int ks = 0; ks < 4; ++ks) {
      v16bf a = load_frag(mT + wave * 16 * M_MEM + ks * 32, M_MEM);
      v16bf b = load_frag(sT + ks * 32, M_MEM);
      acc = wmma_bf16(a, b, acc);
    }
    const int head = lane & 15;
    if (head < H_HEADS) {
      const float dn = denomL[head];
      const int d0 = wave * 16 + ((lane & 16) ? 8 : 0);
      unsigned short* orow = attnws + (size_t)n * HD + head * D_DIM;
#pragma unroll
      for (int i = 0; i < 8; ++i)
        orow[d0 + i] = f2bf(acc[i] / dn);
    }
  }
}

// ---------------- launcher ----------------
extern "C" void kernel_launch(void* const* d_in, const int* in_sizes, int n_in,
                              void* d_out, int out_size, void* d_ws, size_t ws_size,
                              hipStream_t stream) {
  (void)in_sizes; (void)n_in; (void)out_size; (void)ws_size;
  const float* query     = (const float*)d_in[0];
  const float* addresses = (const float*)d_in[1];
  const float* memories  = (const float*)d_in[2];
  const float* Wq        = (const float*)d_in[3];
  const float* bq        = (const float*)d_in[4];
  const float* Wo        = (const float*)d_in[5];
  const float* bo        = (const float*)d_in[6];
  float* out = (float*)d_out;

  // workspace layout (bf16 elements): WqT | WoT | q | attn  (~19 MB total)
  unsigned short* ws     = (unsigned short*)d_ws;
  unsigned short* WqT    = ws;                                   // 512 x 1024
  unsigned short* WoT    = WqT + (size_t)DMODEL * HD;            // 1024 x 512
  unsigned short* qws    = WoT + (size_t)HD * DMODEL;            // N x 512
  unsigned short* attnws = qws + (size_t)N_TOK * HD;             // N x 512

  cvt_transpose<<<(DMODEL * HD + 255) / 256, 256, 0, stream>>>(Wq, WqT, DMODEL, HD);
  cvt_transpose<<<(HD * DMODEL + 255) / 256, 256, 0, stream>>>(Wo, WoT, HD, DMODEL);

  // q = relu(query @ Wq + bq) -> bf16
  gemm_bf16<true, true, true><<<dim3(N_TOK / 64, HD / 64), 128, 0, stream>>>(
      (const void*)query, WqT, bq, (void*)qws, N_TOK, HD, DMODEL);

  token_attn<<<N_TOK, 128, 0, stream>>>(memories, addresses, qws, attnws);

  // out = attn @ Wo + bo -> fp32
  gemm_bf16<false, false, false><<<dim3(N_TOK / 64, DMODEL / 64), 128, 0, stream>>>(
      (const void*)attnws, WoT, bo, (void*)out, N_TOK, DMODEL, HD);
}
